// MLA_8091718385725
// MI455X (gfx1250) — matrix operand order repaired and checked by reading, not compiled, not run on hardware
//
#include <hip/hip_runtime.h>
#include <hip/hip_bf16.h>

typedef __attribute__((ext_vector_type(16))) __bf16 v16bf;
typedef __attribute__((ext_vector_type(8)))  __bf16 v8bf;
typedef __attribute__((ext_vector_type(8)))  float  v8f;
typedef __attribute__((ext_vector_type(4)))  int    v4i;

typedef __attribute__((address_space(1))) v4i GV4i;   // global 16B chunk
typedef __attribute__((address_space(3))) v4i LV4i;   // LDS 16B chunk

// ---------- helpers ----------
__device__ __forceinline__ __bf16 f2bf(float f) { return (__bf16)f; }  // native v_cvt
__device__ __forceinline__ unsigned pack_bf2(float a, float b) {
  unsigned short ua = __builtin_bit_cast(unsigned short, (__bf16)a);
  unsigned short ub = __builtin_bit_cast(unsigned short, (__bf16)b);
  return (unsigned)ua | ((unsigned)ub << 16);
}
// 16B global -> LDS copy; async (ASYNCcnt) when the CDNA5 builtin exists
__device__ __forceinline__ void cp16_g2l(void* ldsDst, const void* gSrc) {
#if __has_builtin(__builtin_amdgcn_global_load_async_to_lds_b128)
  __builtin_amdgcn_global_load_async_to_lds_b128((GV4i*)gSrc, (LV4i*)ldsDst, 0, 0);
#else
  *(v8bf*)ldsDst = *(const v8bf*)gSrc;
#endif
}
__device__ __forceinline__ void cp_wait() {
#if __has_builtin(__builtin_amdgcn_global_load_async_to_lds_b128)
#if __has_builtin(__builtin_amdgcn_s_wait_asynccnt)
  __builtin_amdgcn_s_wait_asynccnt(0);
#else
  asm volatile("s_wait_asynccnt 0" ::: "memory");
#endif
#endif
}
// build a v16bf fragment from two 16-byte runs
__device__ __forceinline__ v16bf ld16(const __bf16* p0, const __bf16* p1) {
  v8bf lo = *(const v8bf*)p0;
  v8bf hi = *(const v8bf*)p1;
  v16bf r;
#pragma unroll
  for (int i = 0; i < 8; ++i) { r[i] = lo[i]; r[i + 8] = hi[i]; }
  return r;
}
__device__ __forceinline__ v8f wmma_bf16(v16bf a, v16bf b, v8f c) {
  return __builtin_amdgcn_wmma_f32_16x16x32_bf16(false, a, false, b, (short)0, c,
                                                 false, false);
}

// ---------- one-time fp32 -> bf16 conversion (flat) ----------
__global__ __launch_bounds__(256) void cvt_f32_to_bf16(
    const float* __restrict__ in, __bf16* __restrict__ out, long n4)
{
  for (long i = (long)blockIdx.x * 256 + threadIdx.x; i < n4;
       i += (long)gridDim.x * 256) {
    float4 v = ((const float4*)in)[i];
    uint2 t; t.x = pack_bf2(v.x, v.y); t.y = pack_bf2(v.z, v.w);
    ((uint2*)out)[i] = t;
  }
}

// ---------- one-time fp32 [K][N] -> bf16 transposed [N][K] ----------
__global__ __launch_bounds__(256) void transpose_cvt_bf16(
    const float* __restrict__ in, __bf16* __restrict__ out, int K, int N)
{
  __shared__ float t[32][33];
  const int kb = blockIdx.y * 32, nb = blockIdx.x * 32;
  const int tx = threadIdx.x & 31, ty = threadIdx.x >> 5;
#pragma unroll
  for (int i = 0; i < 32; i += 8)
    t[ty + i][tx] = in[(size_t)(kb + ty + i) * N + nb + tx];
  __syncthreads();
#pragma unroll
  for (int i = 0; i < 32; i += 8)
    out[(size_t)(nb + ty + i) * K + kb + tx] = f2bf(t[tx][ty + i]);
}

// ---------- bf16-WMMA GEMM: C[M][N](f32) = A[M][K](bf16) * Bt[N][K](bf16) ----------
// block tile 128 x BNT, 8 waves (4 row x 2 col), wave tile 32 x (BNT/2)
template <int BNT>
__global__ __launch_bounds__(256) void gemm_bf16_tpl(
    const __bf16* __restrict__ A, int lda,
    const __bf16* __restrict__ Bt, int ldb,
    float* __restrict__ C, int ldc, int K)
{
  constexpr int TN = BNT / 32;            // 16x16 col-tiles per wave: 2 or 4
  __shared__ __bf16 As[128][40];          // [row][k], 80B rows (16B-aligned)
  __shared__ __bf16 Bs[BNT][40];          // [n][k]

  const int tid = threadIdx.x;
  const int lane = tid & 31;
  const int wid = tid >> 5;
  const int wm = wid >> 1, wn = wid & 1;
  const int l15 = lane & 15, hi = lane >> 4;
  const int rowBase = blockIdx.y * 128;
  const int colBase = blockIdx.x * BNT;

  const v8f vz = {};
  v8f acc[2][TN];
#pragma unroll
  for (int s = 0; s < 2; ++s)
#pragma unroll
    for (int t = 0; t < TN; ++t) acc[s][t] = vz;

  const int akco = hi * 8;
  const int bkco = hi * 8;

  for (int kk = 0; kk < K; kk += 32) {
    __syncthreads();
    // stage A tile: 128x32 bf16 = 512 x 16B chunks, 2/thread (pure copy)
#pragma unroll
    for (int i = 0; i < 2; ++i) {
      int idx = tid + i * 256;
      int r = idx >> 2, c = idx & 3;
      cp16_g2l(&As[r][c * 8], A + (size_t)(rowBase + r) * lda + kk + c * 8);
    }
    // stage B tile: BNT x 32 bf16 (weights pre-transposed -> contiguous copy)
#pragma unroll
    for (int i = 0; i < BNT / 64; ++i) {
      int idx = tid + i * 256;
      int n = idx >> 2, c = idx & 3;
      cp16_g2l(&Bs[n][c * 8], Bt + (size_t)(colBase + n) * ldb + kk + c * 8);
    }
    if (kk + 32 < K) {  // global_prefetch_b8 for the next K tile
      __builtin_prefetch(A + (size_t)(rowBase + (tid >> 1)) * lda + kk + 32, 0, 1);
      __builtin_prefetch(Bt + (size_t)(colBase + (tid & (BNT - 1))) * ldb + kk + 32, 0, 1);
    }
    cp_wait();
    __syncthreads();

    v16bf af[2], bf[TN];
#pragma unroll
    for (int s = 0; s < 2; ++s) {
      const __bf16* p = &As[wm * 32 + s * 16 + l15][0];
      af[s] = ld16(p + akco, p + akco + 16);
    }
#pragma unroll
    for (int t = 0; t < TN; ++t) {
      const __bf16* p = &Bs[wn * (BNT / 2) + t * 16 + l15][0];
      bf[t] = ld16(p + bkco, p + bkco + 16);
    }
#pragma unroll
    for (int s = 0; s < 2; ++s)
#pragma unroll
      for (int t = 0; t < TN; ++t)
        acc[s][t] = wmma_bf16(af[s], bf[t], acc[s][t]);
  }

#pragma unroll
  for (int s = 0; s < 2; ++s)
#pragma unroll
    for (int t = 0; t < TN; ++t)
#pragma unroll
      for (int r = 0; r < 8; ++r) {
        int row = rowBase + wm * 32 + s * 16 + hi * 8 + r;
        int col = colBase + wn * (BNT / 2) + t * 16 + l15;
        C[(size_t)row * ldc + col] = acc[s][t][r];
      }
}

// ---------- row RMSNorm: f32 in -> bf16 out ----------
__global__ __launch_bounds__(256) void rmsnorm_bf16(
    const float* __restrict__ X, const float* __restrict__ w,
    __bf16* __restrict__ Y, int C, int strideX, int colOff, int ldy)
{
  __shared__ float red[8];
  const float* xr = X + (size_t)blockIdx.x * strideX + colOff;
  __bf16* yr = Y + (size_t)blockIdx.x * ldy;
  float ss = 0.f;
  for (int c = threadIdx.x; c < C; c += 256) { float v = xr[c]; ss += v * v; }
#pragma unroll
  for (int m = 16; m >= 1; m >>= 1) ss += __shfl_xor(ss, m, 32);
  if ((threadIdx.x & 31) == 0) red[threadIdx.x >> 5] = ss;
  __syncthreads();
  if (threadIdx.x == 0) {
    float t = 0.f;
#pragma unroll
    for (int i = 0; i < 8; ++i) t += red[i];
    red[0] = rsqrtf(t / (float)C + 1e-6f);
  }
  __syncthreads();
  float sc = red[0];
  for (int c = threadIdx.x; c < C; c += 256) yr[c] = f2bf(xr[c] * sc * w[c]);
}

// ---------- RoPE + K/V assembly -> bf16 layouts for attention ----------
__global__ __launch_bounds__(256) void prepare_qkv(
    const float* __restrict__ qf, const float* __restrict__ kvf,
    const float* __restrict__ kva,
    const float* __restrict__ cosT, const float* __restrict__ sinT,
    __bf16* __restrict__ qb, __bf16* __restrict__ kb, __bf16* __restrict__ vt,
    int Nn)
{
  const int tok = blockIdx.x;
  const int b = tok / Nn, n = tok % Nn;
  const float* cr = cosT + (size_t)n * 32;
  const float* sr = sinT + (size_t)n * 32;

  for (int idx = threadIdx.x; idx < 16 * 192; idx += 256) {
    int h = idx / 192, d = idx % 192;
    size_t dst = (((size_t)(b * 16 + h)) * Nn + n) * 192 + d;
    float qv, kv;
    if (d < 128) {
      qv = qf[(size_t)tok * 3072 + h * 192 + d];
      kv = kvf[(size_t)tok * 4096 + h * 256 + d];
    } else {
      int j = (d - 128) >> 1, odd = (d - 128) & 1;
      float c = cr[j], s = sr[j];
      float q0 = qf[(size_t)tok * 3072 + h * 192 + 128 + 2 * j];
      float q1 = qf[(size_t)tok * 3072 + h * 192 + 128 + 2 * j + 1];
      qv = odd ? (q0 * s + q1 * c) : (q0 * c - q1 * s);
      float k0 = kva[(size_t)tok * 576 + 2 * j];
      float k1 = kva[(size_t)tok * 576 + 2 * j + 1];
      kv = odd ? (k0 * s + k1 * c) : (k0 * c - k1 * s);
    }
    qb[dst] = f2bf(qv);
    kb[dst] = f2bf(kv);
  }
  for (int idx = threadIdx.x; idx < 16 * 128; idx += 256) {
    int h = idx / 128, dv = idx % 128;
    float v = kvf[(size_t)tok * 4096 + h * 256 + 128 + dv];
    vt[(((size_t)(b * 16 + h)) * 128 + dv) * Nn + n] = f2bf(v);
  }
}

// ---------- flash attention (causal), 4 waves x 16 q-rows per block ----------
__global__ __launch_bounds__(128) void mla_attention(
    const __bf16* __restrict__ qb,  // [B*H][N][192]
    const __bf16* __restrict__ kb,  // [B*H][N][192]
    const __bf16* __restrict__ vt,  // [B*H][128][N]
    __bf16* __restrict__ out,       // [B*N][2048] bf16 (h*128+dv)
    int Nn)
{
  __shared__ __bf16 Kt[32][200];
  __shared__ __bf16 Vt[128][40];
  __shared__ __bf16 Ps[4][16][32];

  const int bh = blockIdx.y;
  const int tid = threadIdx.x;
  const int lane = tid & 31, wid = tid >> 5;
  const int l15 = lane & 15, hi = lane >> 4;
  const int qrow0 = blockIdx.x * 64 + wid * 16;
  const int akco = hi * 8, bkco = hi * 8;

  const __bf16* qbase = qb + ((size_t)bh * Nn + qrow0 + l15) * 192;
  v16bf qfrag[6];
#pragma unroll
  for (int d = 0; d < 6; ++d)
    qfrag[d] = ld16(qbase + d * 32 + akco, qbase + d * 32 + akco + 16);

  const v8f vz = {};
  v8f o[8] = {vz, vz, vz, vz, vz, vz, vz, vz};
  float m[8], l[8];
#pragma unroll
  for (int r = 0; r < 8; ++r) { m[r] = -1e30f; l[r] = 0.f; }

  const float scale = 0.07216878364870323f;  // 192^-0.5
  const int nkb = 2 * (blockIdx.x + 1);

  for (int kbi = 0; kbi < nkb; ++kbi) {
    const int k0 = kbi * 32;
    __syncthreads();
#pragma unroll
    for (int i = 0; i < 6; ++i) {  // K tile: 32x192 bf16, 768 chunks
      int idx = tid + i * 128;
      int kr = idx / 24, c = idx % 24;
      cp16_g2l(&Kt[kr][c * 8], kb + ((size_t)bh * Nn + k0 + kr) * 192 + c * 8);
    }
#pragma unroll
    for (int i = 0; i < 4; ++i) {  // V^T tile: 128x32 bf16, 512 chunks
      int idx = tid + i * 128;
      int dr = idx >> 2, c = idx & 3;
      cp16_g2l(&Vt[dr][c * 8], vt + ((size_t)bh * 128 + dr) * Nn + k0 + c * 8);
    }
    cp_wait();
    __syncthreads();

    v8f s0 = {}, s1 = {};
#pragma unroll
    for (int d = 0; d < 6; ++d) {
      const __bf16* kp0 = &Kt[l15][d * 32 + bkco];
      s0 = wmma_bf16(qfrag[d], ld16(kp0, kp0 + 16), s0);
      const __bf16* kp1 = &Kt[16 + l15][d * 32 + bkco];
      s1 = wmma_bf16(qfrag[d], ld16(kp1, kp1 + 16), s1);
    }

#pragma unroll
    for (int r = 0; r < 8; ++r) {
      int row = qrow0 + hi * 8 + r;
      int key0 = k0 + l15, key1 = key0 + 16;
      float a = s0[r] * scale + (key0 > row ? -1e30f : 0.f);
      float b = s1[r] * scale + (key1 > row ? -1e30f : 0.f);
      float mx = fmaxf(a, b);
      mx = fmaxf(mx, __shfl_xor(mx, 1, 32));
      mx = fmaxf(mx, __shfl_xor(mx, 2, 32));
      mx = fmaxf(mx, __shfl_xor(mx, 4, 32));
      mx = fmaxf(mx, __shfl_xor(mx, 8, 32));
      float mn = fmaxf(m[r], mx);
      float alpha = __expf(m[r] - mn);
      float p0 = __expf(a - mn);
      float p1 = __expf(b - mn);
      float rs = p0 + p1;
      rs += __shfl_xor(rs, 1, 32);
      rs += __shfl_xor(rs, 2, 32);
      rs += __shfl_xor(rs, 4, 32);
      rs += __shfl_xor(rs, 8, 32);
      l[r] = l[r] * alpha + rs;
      m[r] = mn;
#pragma unroll
      for (int c = 0; c < 8; ++c) o[c][r] *= alpha;
      Ps[wid][hi * 8 + r][l15]      = f2bf(p0);
      Ps[wid][hi * 8 + r][16 + l15] = f2bf(p1);
    }

    const __bf16* pp = &Ps[wid][l15][akco];
    v16bf pf = ld16(pp, pp + 16);
#pragma unroll
    for (int c = 0; c < 8; ++c) {
      const __bf16* vp = &Vt[c * 16 + l15][bkco];
      o[c] = wmma_bf16(pf, ld16(vp, vp + 16), o[c]);
    }
  }

  const int b = bh >> 4, h = bh & 15;
#pragma unroll
  for (int c = 0; c < 8; ++c)
#pragma unroll
    for (int r = 0; r < 8; ++r) {
      int row = qrow0 + hi * 8 + r;
      out[((size_t)(b * Nn + row)) * 2048 + h * 128 + c * 16 + l15] =
          f2bf(o[c][r] / l[r]);
    }
}

// ---------- host launcher ----------
extern "C" void kernel_launch(void* const* d_in, const int* in_sizes, int n_in,
                              void* d_out, int out_size, void* d_ws, size_t ws_size,
                              hipStream_t stream) {
  (void)in_sizes; (void)n_in; (void)out_size; (void)ws_size;
  const float* x     = (const float*)d_in[0];
  const float* cosT  = (const float*)d_in[2];
  const float* sinT  = (const float*)d_in[3];
  const float* wq_a  = (const float*)d_in[4];
  const float* q_nw  = (const float*)d_in[5];
  const float* wq_b  = (const float*)d_in[6];
  const float* wkv_a = (const float*)d_in[7];
  const float* kv_nw = (const float*)d_in[8];
  const float* wkv_b = (const float*)d_in[9];
  const float* w_out = (const float*)d_in[10];
  float* out = (float*)d_out;

  char* ws = (char*)d_ws;
  __bf16* xb     = (__bf16*)(ws + 0);           // [4096][2048]
  __bf16* wqab   = (__bf16*)(ws + 16777216);    // [1536][2048]  (transposed)
  __bf16* wqbb   = (__bf16*)(ws + 23068672);    // [3072][1536]
  __bf16* wkvab  = (__bf16*)(ws + 32505856);    // [576][2048]
  __bf16* wkvbb  = (__bf16*)(ws + 34865152);    // [4096][512]
  __bf16* woutb  = (__bf16*)(ws + 39059456);    // [2048][2048]
  float*  q_a    = (float*)(ws + 47448064);     // [4096][1536] f32
  __bf16* q_abf  = (__bf16*)(ws + 72613888);    // [4096][1536]
  float*  q_full = (float*)(ws + 85196800);     // [4096][3072] f32
  float*  kv_a   = (float*)(ws + 135528448);    // [4096][576]  f32
  __bf16* kvcbf  = (__bf16*)(ws + 144965632);   // [4096][512]
  float*  kv_ful = (float*)(ws + 149159936);    // [4096][4096] f32
  __bf16* kb16   = (__bf16*)(ws + 216268800);   // [32][2048][192]
  __bf16* vt16   = (__bf16*)(ws + 241434624);   // [32][128][2048]
  __bf16* qb16   = (__bf16*)(ws + 47448064);    // reuse dead q_a f32 region
  __bf16* attnbf = (__bf16*)(ws + 149159936);   // reuse dead kv_full region

  const int NT = 2048;
  const dim3 blk256(256);

  // one-time conversions / weight transposes
  cvt_f32_to_bf16<<<2048, blk256, 0, stream>>>(x, xb, 4096L * 2048 / 4);
  transpose_cvt_bf16<<<dim3(1536 / 32, 2048 / 32), blk256, 0, stream>>>(wq_a, wqab, 2048, 1536);
  transpose_cvt_bf16<<<dim3(3072 / 32, 1536 / 32), blk256, 0, stream>>>(wq_b, wqbb, 1536, 3072);
  transpose_cvt_bf16<<<dim3(576 / 32, 2048 / 32), blk256, 0, stream>>>(wkv_a, wkvab, 2048, 576);
  transpose_cvt_bf16<<<dim3(4096 / 32, 512 / 32), blk256, 0, stream>>>(wkv_b, wkvbb, 512, 4096);
  transpose_cvt_bf16<<<dim3(2048 / 32, 2048 / 32), blk256, 0, stream>>>(w_out, woutb, 2048, 2048);

  // q_a = x @ wq_a ; rmsnorm -> bf16
  gemm_bf16_tpl<128><<<dim3(1536 / 128, 32), blk256, 0, stream>>>(xb, 2048, wqab, 2048, q_a, 1536, 2048);
  rmsnorm_bf16<<<4096, blk256, 0, stream>>>(q_a, q_nw, q_abf, 1536, 1536, 0, 1536);
  // q = q_a @ wq_b
  gemm_bf16_tpl<128><<<dim3(3072 / 128, 32), blk256, 0, stream>>>(q_abf, 1536, wqbb, 1536, q_full, 3072, 1536);
  // kv_a = x @ wkv_a  (N=576 -> 64-wide tiles)
  gemm_bf16_tpl<64><<<dim3(576 / 64, 32), blk256, 0, stream>>>(xb, 2048, wkvab, 2048, kv_a, 576, 2048);
  rmsnorm_bf16<<<4096, blk256, 0, stream>>>(kv_a, kv_nw, kvcbf, 512, 576, 64, 512);
  // kv = kv_c @ wkv_b
  gemm_bf16_tpl<128><<<dim3(4096 / 128, 32), blk256, 0, stream>>>(kvcbf, 512, wkvbb, 512, kv_ful, 4096, 512);
  // RoPE + bf16 attention layouts
  prepare_qkv<<<4096, blk256, 0, stream>>>(q_full, kv_ful, kv_a, cosT, sinT, qb16, kb16, vt16, NT);
  // causal flash attention -> bf16
  mla_attention<<<dim3(NT / 64, 32), dim3(128), 0, stream>>>(qb16, kb16, vt16, attnbf, NT);
  // out = attn @ w_out
  gemm_bf16_tpl<128><<<dim3(2048 / 128, 32), blk256, 0, stream>>>(attnbf, 2048, woutb, 2048, out, 2048, 2048);
}
